// Block_63067299774517
// MI455X (gfx1250) — matrix operand order repaired and checked by reading, not compiled, not run on hardware
//
#include <hip/hip_runtime.h>
#include <hip/hip_bf16.h>

// Problem constants (B=2, T=2048, C=1024, H=16, HD=64)
#define TSEQ 2048
#define CDIM 1024
#define NHEAD 16
#define MROWS 4096   // B*T

typedef __attribute__((ext_vector_type(16))) __bf16 bf16x16;
typedef __attribute__((ext_vector_type(8)))  __bf16 bf16x8;
typedef __attribute__((ext_vector_type(8)))  float  f32x8;
typedef unsigned int u32x4 __attribute__((ext_vector_type(4)));
typedef int          i32x4v __attribute__((ext_vector_type(4)));
typedef int          i32x8v __attribute__((ext_vector_type(8)));

// ---- Tensor Data Mover availability (ROCm7.2 = 5-arg, therock = 6-arg) ----
#if __has_builtin(__builtin_amdgcn_tensor_load_to_lds)
#  if __has_include(<hip/amd_detail/amd_gfx1250_TDM.h>)
#    define TDM_6ARG 1
#  else
#    define TDM_5ARG 1
#  endif
#  define USE_TDM 1
#endif

union Frag { bf16x16 v; bf16x8 h[2]; };

// A-matrix 16x32 bf16 fragment from LDS, row-major source with given element stride.
// Lanes 0-15: row=lane, K={kb+0..7, kb+16..23}; lanes 16-31: row=lane-16, K={kb+8..15, kb+24..31}.
__device__ __forceinline__ Frag lds_load_a(const __bf16* base, int row, int stride, int kbase) {
  int lane = threadIdx.x & 31;
  const __bf16* p = base + (size_t)(row + (lane & 15)) * stride + kbase + ((lane >> 4) << 3);
  Frag f;
  f.h[0] = *reinterpret_cast<const bf16x8*>(p);
  f.h[1] = *reinterpret_cast<const bf16x8*>(p + 16);
  return f;
}

// B-matrix 32x16 bf16 fragment from LDS stored transposed [col][k].
__device__ __forceinline__ Frag lds_load_bt(const __bf16* baseT, int col, int stride, int kbase) {
  int lane = threadIdx.x & 31;
  const __bf16* p = baseT + (size_t)(col + (lane & 15)) * stride + kbase + ((lane >> 4) << 3);
  Frag f;
  f.h[0] = *reinterpret_cast<const bf16x8*>(p);
  f.h[1] = *reinterpret_cast<const bf16x8*>(p + 16);
  return f;
}

__device__ __forceinline__ f32x8 wmma_bf16(Frag a, Frag b, f32x8 c) {
  return __builtin_amdgcn_wmma_f32_16x16x32_bf16(
      false, a.v, false, b.v, (short)0, c, false, false);
}

__device__ __forceinline__ void cp16(__bf16* dst, const __bf16* src) {
  *reinterpret_cast<bf16x8*>(dst)     = *reinterpret_cast<const bf16x8*>(src);
  *reinterpret_cast<bf16x8*>(dst + 8) = *reinterpret_cast<const bf16x8*>(src + 8);
}

// LDS byte offset of a generic pointer to shared memory (aperture: addr[31:0]).
__device__ __forceinline__ unsigned lds_off(const void* p) {
  return (unsigned)(unsigned long long)(uintptr_t)p;
}

// TDM 2-D bf16 tile load: tile_d0 contiguous elements per row, tile_d1 rows,
// row stride `stride` elements. Descriptor per CDNA5 ISA.
__device__ __forceinline__ void tdm_load_bf16_2d(const void* gsrc, unsigned ldsoff,
                                                 int tile_d0, int tile_d1, int stride) {
#ifdef USE_TDM
  unsigned long long ga = (unsigned long long)(uintptr_t)gsrc;
  u32x4 g0;
  g0.x = 1u;                               // count=1, user mode
  g0.y = ldsoff;                           // lds_addr [63:32]
  g0.z = (unsigned)ga;                     // global_addr low  [95:64]
  g0.w = (unsigned)(ga >> 32) | (2u << 30);// global_addr high + type=2 [127:126]
  i32x8v g1;
  g1[0] = 1 << 16;                         // data_size=1 (2 bytes), mask=0
  g1[1] = 0;                               // tensor_dim0 = 1<<30 (low16 = 0)
  g1[2] = 0x4000;                          // tensor_dim0 hi16; tensor_dim1=1<<20 (low16=0)
  g1[3] = 0x10 | (tile_d0 << 16);          // tensor_dim1 hi16; tile_dim0
  g1[4] = tile_d1 & 0xFFFF;                // tile_dim1; tile_dim2=0
  g1[5] = stride;                          // tensor_dim0_stride low32
  g1[6] = 0;                               // stride hi16; dim1 stride low16
  g1[7] = 0;
  i32x4v gz = {0, 0, 0, 0};
#ifdef TDM_6ARG
  i32x8v gz8 = {0, 0, 0, 0, 0, 0, 0, 0};
  __builtin_amdgcn_tensor_load_to_lds(g0, g1, gz, gz, gz8, 0);
#else
  __builtin_amdgcn_tensor_load_to_lds(g0, g1, gz, gz, 0);
#endif
#else
  (void)gsrc; (void)ldsoff; (void)tile_d0; (void)tile_d1; (void)stride;
#endif
}

// ---------------------------------------------------------------------------
// LayerNorm(fp32) -> bf16. One block per row, 256 threads, C=1024.
// ---------------------------------------------------------------------------
__global__ __launch_bounds__(256) void ln_to_bf16(const float* __restrict__ x,
                                                  const float* __restrict__ g,
                                                  const float* __restrict__ be,
                                                  __bf16* __restrict__ out) {
  __shared__ float red[256];
  int row = blockIdx.x, tid = threadIdx.x;
  const float* xr = x + (size_t)row * CDIM;
  float v[4];
  float s = 0.f;
#pragma unroll
  for (int i = 0; i < 4; ++i) { v[i] = xr[tid + i * 256]; s += v[i]; }
  red[tid] = s; __syncthreads();
  for (int st = 128; st > 0; st >>= 1) { if (tid < st) red[tid] += red[tid + st]; __syncthreads(); }
  float mu = red[0] * (1.f / CDIM);
  __syncthreads();
  s = 0.f;
#pragma unroll
  for (int i = 0; i < 4; ++i) { float d = v[i] - mu; s += d * d; }
  red[tid] = s; __syncthreads();
  for (int st = 128; st > 0; st >>= 1) { if (tid < st) red[tid] += red[tid + st]; __syncthreads(); }
  float rs = rsqrtf(red[0] * (1.f / CDIM) + 1e-5f);
#pragma unroll
  for (int i = 0; i < 4; ++i) {
    int c = tid + i * 256;
    out[(size_t)row * CDIM + c] = (__bf16)((v[i] - mu) * rs * g[c] + be[c]);
  }
}

// ---------------------------------------------------------------------------
// Weight prep
// ---------------------------------------------------------------------------
__global__ __launch_bounds__(256) void pack_qkv_w(const float* __restrict__ Wq,
                                                  const float* __restrict__ Wk,
                                                  const float* __restrict__ Wv,
                                                  __bf16* __restrict__ out) {
  int idx = blockIdx.x * 256 + threadIdx.x;   // exactly C*3C threads
  int c = idx / 3072, n = idx % 3072;
  int sel = n >> 10, w = n & 1023, hh = w >> 6, d = w & 63;
  const float* W = (sel == 0) ? Wq : (sel == 1) ? Wk : Wv;
  out[idx] = (__bf16)W[((size_t)hh * CDIM + c) * 64 + d];
}

__global__ __launch_bounds__(256) void f32_to_bf16(const float* __restrict__ in,
                                                   __bf16* __restrict__ out, int n) {
  int idx = blockIdx.x * 256 + threadIdx.x;
  if (idx < n) out[idx] = (__bf16)in[idx];
}

// ---------------------------------------------------------------------------
// bf16 WMMA GEMM, tile 128x128, BK=64, double-buffered LDS (64KB dynamic).
// A tile staged by the Tensor Data Mover (wave 0), B tile staged transposed by
// all waves; one barrier per K-step; 16 WMMAs per stage per wave.
// LDS layout per buffer (16384 elems): [0,8192) = A [128][64], [8192,16384) = B^T [128][64].
// ---------------------------------------------------------------------------
__global__ __launch_bounds__(256) void gemm_bf16(const __bf16* __restrict__ A,
                                                 const __bf16* __restrict__ Bm,
                                                 int N, int K,
                                                 const float* __restrict__ bias,
                                                 const float* __restrict__ resid,
                                                 float* __restrict__ outF,
                                                 __bf16* __restrict__ outB,
                                                 int relu) {
  extern __shared__ char gsm[];

  int tid = threadIdx.x, lane = tid & 31, wave = tid >> 5;
  int wm = wave & 3, wn = wave >> 2;
  int bm0 = blockIdx.y * 128, bn0 = blockIdx.x * 128;
  f32x8 zero = {0.f, 0.f, 0.f, 0.f, 0.f, 0.f, 0.f, 0.f};
  f32x8 acc[2][4];
#pragma unroll
  for (int i = 0; i < 2; ++i)
#pragma unroll
    for (int j = 0; j < 4; ++j) acc[i][j] = zero;

  // B staging decomposition: 64x128 tile, 32 elems/thread
  int bkk = tid >> 2, bnseg = (tid & 3) * 32;
#ifndef USE_TDM
  int arow = tid >> 1, aseg = (tid & 1) * 32;
#endif

  const int nk = K >> 6;

  auto stage = [&](int i, int buf) {
    int k0 = i << 6;
    __bf16* asb = (__bf16*)gsm + buf * 16384;        // A tile [128][64]
    __bf16* bsb = asb + 8192;                        // B^T tile [128][64]
    // ---- A tile via TDM (row-major [128][64]) ----
#ifdef USE_TDM
    if (wave == 0)
      tdm_load_bf16_2d(A + (size_t)bm0 * K + k0, lds_off(asb), 64, 128, K);
#else
    cp16(&asb[arow * 64 + aseg],      &A[(size_t)(bm0 + arow) * K + k0 + aseg]);
    cp16(&asb[arow * 64 + aseg + 16], &A[(size_t)(bm0 + arow) * K + k0 + aseg + 16]);
#endif
    // ---- B tile transposed [n][k] ----
    const __bf16* src = &Bm[(size_t)(k0 + bkk) * N + bn0 + bnseg];
    __bf16 tmp[32];
    *reinterpret_cast<bf16x8*>(tmp)      = *reinterpret_cast<const bf16x8*>(src);
    *reinterpret_cast<bf16x8*>(tmp + 8)  = *reinterpret_cast<const bf16x8*>(src + 8);
    *reinterpret_cast<bf16x8*>(tmp + 16) = *reinterpret_cast<const bf16x8*>(src + 16);
    *reinterpret_cast<bf16x8*>(tmp + 24) = *reinterpret_cast<const bf16x8*>(src + 24);
#pragma unroll
    for (int t = 0; t < 32; ++t) bsb[(bnseg + t) * 64 + bkk] = tmp[t];
    if (k0 + 64 < K)  // emits global_prefetch_b8 for the tile after this one
      __builtin_prefetch(&Bm[(size_t)(k0 + 64 + bkk) * N + bn0 + bnseg], 0, 1);
  };

  // prologue
  stage(0, 0);
#ifdef USE_TDM
  if (wave == 0) __builtin_amdgcn_s_wait_tensorcnt(0);
#endif
  __syncthreads();

  for (int i = 0; i < nk; ++i) {
    int cur = i & 1;
    if (i + 1 < nk) stage(i + 1, cur ^ 1);
    const __bf16* as = (const __bf16*)gsm + cur * 16384;
    const __bf16* bs = as + 8192;
#pragma unroll
    for (int ks = 0; ks < 2; ++ks) {
      Frag af0 = lds_load_a(as, wm * 32,      64, ks * 32);
      Frag af1 = lds_load_a(as, wm * 32 + 16, 64, ks * 32);
#pragma unroll
      for (int tn = 0; tn < 4; ++tn) {
        Frag bf = lds_load_bt(bs, wn * 64 + tn * 16, 64, ks * 32);
        acc[0][tn] = wmma_bf16(af0, bf, acc[0][tn]);
        acc[1][tn] = wmma_bf16(af1, bf, acc[1][tn]);
      }
    }
#ifdef USE_TDM
    if (wave == 0) __builtin_amdgcn_s_wait_tensorcnt(0);
#endif
    __syncthreads();
  }

  int hi = lane >> 4, ln = lane & 15;
#pragma unroll
  for (int tm = 0; tm < 2; ++tm)
#pragma unroll
    for (int tn = 0; tn < 4; ++tn)
#pragma unroll
      for (int j = 0; j < 8; ++j) {
        int row = bm0 + wm * 32 + tm * 16 + j + hi * 8;
        int col = bn0 + wn * 64 + tn * 16 + ln;
        float v = acc[tm][tn][j];
        if (bias)  v += bias[col];
        if (relu)  v = fmaxf(v, 0.f);
        if (resid) v += resid[(size_t)row * N + col];
        if (outF)  outF[(size_t)row * N + col] = v;
        if (outB)  outB[(size_t)row * N + col] = (__bf16)v;
      }
}

// ---------------------------------------------------------------------------
// Causal flash attention. Grid: (T/128, B*H). qkv = [M, 3072] bf16.
// ---------------------------------------------------------------------------
__global__ __launch_bounds__(256) void attn_kernel(const __bf16* __restrict__ qkv,
                                                   __bf16* __restrict__ o) {
  extern __shared__ char smem[];
  __bf16* qs  = (__bf16*)smem;               // [128][64]
  __bf16* ks  = qs + 128 * 64;               // [128][64]  (== transposed-B for Q*K^T)
  __bf16* vts = ks + 128 * 64;               // [64][128]  V transposed
  float*  sS  = (float*)(vts + 64 * 128);    // [128][128] scores; P bf16 written in-place
  float*  mrow = sS + 128 * 128;
  float*  lrow = mrow + 128;
  float*  frow = lrow + 128;

  int tid = threadIdx.x, lane = tid & 31, wave = tid >> 5;
  int hi = lane >> 4, ln = lane & 15;
  int qt = blockIdx.x;
  int bh = blockIdx.y;
  int b = bh >> 4, h = bh & 15;
  size_t qrow0 = (size_t)b * TSEQ + (size_t)qt * 128;

  {
    int r = tid >> 1, seg = (tid & 1) * 32;
    const __bf16* src = qkv + (qrow0 + r) * 3072 + h * 64 + seg;
    cp16(&qs[r * 64 + seg], src);
    cp16(&qs[r * 64 + seg + 16], src + 16);
  }
  if (tid < 128) { mrow[tid] = -3.0e38f; lrow[tid] = 0.f; }

  f32x8 zero = {0.f, 0.f, 0.f, 0.f, 0.f, 0.f, 0.f, 0.f};
  f32x8 oacc[4];
#pragma unroll
  for (int i = 0; i < 4; ++i) oacc[i] = zero;
  __syncthreads();

  int wm = wave & 3, wn = wave >> 2;

  for (int kt = 0; kt <= qt; ++kt) {
    size_t krow0 = (size_t)b * TSEQ + (size_t)kt * 128;
    {
      int r = tid >> 1, seg = (tid & 1) * 32;
      const __bf16* ksrc = qkv + (krow0 + r) * 3072 + 1024 + h * 64 + seg;
      cp16(&ks[r * 64 + seg], ksrc);
      cp16(&ks[r * 64 + seg + 16], ksrc + 16);
      const __bf16* vsrc = qkv + (krow0 + r) * 3072 + 2048 + h * 64 + seg;
      __bf16 tmp[32];
      *reinterpret_cast<bf16x8*>(tmp)      = *reinterpret_cast<const bf16x8*>(vsrc);
      *reinterpret_cast<bf16x8*>(tmp + 8)  = *reinterpret_cast<const bf16x8*>(vsrc + 8);
      *reinterpret_cast<bf16x8*>(tmp + 16) = *reinterpret_cast<const bf16x8*>(vsrc + 16);
      *reinterpret_cast<bf16x8*>(tmp + 24) = *reinterpret_cast<const bf16x8*>(vsrc + 24);
#pragma unroll
      for (int i = 0; i < 32; ++i) vts[(seg + i) * 128 + r] = tmp[i];
    }
    __syncthreads();

    f32x8 sacc[2][4];
#pragma unroll
    for (int i = 0; i < 2; ++i)
#pragma unroll
      for (int j = 0; j < 4; ++j) sacc[i][j] = zero;
#pragma unroll
    for (int kd = 0; kd < 2; ++kd) {
      Frag af[2], bfv[4];
#pragma unroll
      for (int tm = 0; tm < 2; ++tm) af[tm] = lds_load_a(qs, wm * 32 + tm * 16, 64, kd * 32);
#pragma unroll
      for (int tn = 0; tn < 4; ++tn) bfv[tn] = lds_load_bt(ks, wn * 64 + tn * 16, 64, kd * 32);
#pragma unroll
      for (int tm = 0; tm < 2; ++tm)
#pragma unroll
        for (int tn = 0; tn < 4; ++tn)
          sacc[tm][tn] = wmma_bf16(af[tm], bfv[tn], sacc[tm][tn]);
    }
#pragma unroll
    for (int tm = 0; tm < 2; ++tm)
#pragma unroll
      for (int tn = 0; tn < 4; ++tn)
#pragma unroll
        for (int j = 0; j < 8; ++j) {
          int row = wm * 32 + tm * 16 + j + hi * 8;
          int col = wn * 64 + tn * 16 + ln;
          float v = sacc[tm][tn][j] * 0.125f;      // HD^-0.5
          if (kt * 128 + col > qt * 128 + row) v = -3.0e38f;
          sS[row * 128 + col] = v;
        }
    __syncthreads();

    if (tid < 128) {
      int r = tid;
      float mold = mrow[r], mnew = mold;
      for (int j = 0; j < 128; ++j) mnew = fmaxf(mnew, sS[r * 128 + j]);
      float fac = __expf(mold - mnew);
      float l = lrow[r] * fac;
      __bf16* prow = (__bf16*)&sS[r * 128];
      for (int j = 0; j < 128; ++j) {
        float p = __expf(sS[r * 128 + j] - mnew);
        l += p;
        prow[j] = (__bf16)p;   // in-place f32->bf16: write 2j < later reads 4j'
      }
      mrow[r] = mnew; lrow[r] = l; frow[r] = fac;
    }
    __syncthreads();

    int rb = wave * 16;
#pragma unroll
    for (int t = 0; t < 4; ++t)
#pragma unroll
      for (int j = 0; j < 8; ++j) oacc[t][j] *= frow[rb + j + hi * 8];
#pragma unroll
    for (int kd = 0; kd < 4; ++kd) {
      Frag pa = lds_load_a((const __bf16*)sS, rb, 256, kd * 32);  // P row stride 256 bf16
#pragma unroll
      for (int t = 0; t < 4; ++t) {
        Frag vb = lds_load_bt(vts, t * 16, 128, kd * 32);
        oacc[t] = wmma_bf16(pa, vb, oacc[t]);
      }
    }
    __syncthreads();
  }

  int rb = wave * 16;
#pragma unroll
  for (int t = 0; t < 4; ++t)
#pragma unroll
    for (int j = 0; j < 8; ++j) {
      int row = rb + j + hi * 8;
      float v = oacc[t][j] / lrow[row];
      o[(qrow0 + row) * CDIM + h * 64 + t * 16 + ln] = (__bf16)v;
    }
}

// ---------------------------------------------------------------------------
extern "C" void kernel_launch(void* const* d_in, const int* in_sizes, int n_in,
                              void* d_out, int out_size, void* d_ws, size_t ws_size,
                              hipStream_t stream) {
  const float* x     = (const float*)d_in[0];
  const float* Wq    = (const float*)d_in[1];
  const float* Wk    = (const float*)d_in[2];
  const float* Wv    = (const float*)d_in[3];
  const float* Wproj = (const float*)d_in[4];
  const float* bproj = (const float*)d_in[5];
  const float* W1    = (const float*)d_in[6];
  const float* b1    = (const float*)d_in[7];
  const float* W2    = (const float*)d_in[8];
  const float* b2    = (const float*)d_in[9];
  const float* g1    = (const float*)d_in[10];
  const float* be1   = (const float*)d_in[11];
  const float* g2    = (const float*)d_in[12];
  const float* be2   = (const float*)d_in[13];
  (void)in_sizes; (void)n_in; (void)out_size; (void)ws_size;

  char* ws = (char*)d_ws;
  size_t off = 0;
  auto carve = [&](size_t bytes) {
    void* p = ws + off;
    off += (bytes + 255) & ~(size_t)255;
    return p;
  };
  __bf16* h1    = (__bf16*)carve((size_t)MROWS * CDIM * 2);
  __bf16* wqkv  = (__bf16*)carve((size_t)CDIM * 3072 * 2);
  __bf16* qkv   = (__bf16*)carve((size_t)MROWS * 3072 * 2);
  __bf16* obf   = (__bf16*)carve((size_t)MROWS * CDIM * 2);
  __bf16* wproj = (__bf16*)carve((size_t)CDIM * CDIM * 2);
  float*  x1    = (float*)carve((size_t)MROWS * CDIM * 4);
  __bf16* h2    = (__bf16*)carve((size_t)MROWS * CDIM * 2);
  __bf16* w1b   = (__bf16*)carve((size_t)CDIM * 4096 * 2);
  __bf16* ff1   = (__bf16*)carve((size_t)MROWS * 4096 * 2);
  __bf16* w2b   = (__bf16*)carve((size_t)4096 * CDIM * 2);

  const size_t gemm_smem = 4 * 8192 * sizeof(__bf16);  // 64 KB, double-buffered

  // 1) LN1
  ln_to_bf16<<<MROWS, 256, 0, stream>>>(x, g1, be1, h1);
  // 2) weight prep
  pack_qkv_w<<<(CDIM * 3072) / 256, 256, 0, stream>>>(Wq, Wk, Wv, wqkv);
  f32_to_bf16<<<(CDIM * CDIM) / 256, 256, 0, stream>>>(Wproj, wproj, CDIM * CDIM);
  f32_to_bf16<<<(CDIM * 4096) / 256, 256, 0, stream>>>(W1, w1b, CDIM * 4096);
  f32_to_bf16<<<(4096 * CDIM) / 256, 256, 0, stream>>>(W2, w2b, 4096 * CDIM);
  // 3) QKV = h1 @ Wqkv -> bf16 [M,3072]
  gemm_bf16<<<dim3(3072 / 128, MROWS / 128), 256, gemm_smem, stream>>>(
      h1, wqkv, 3072, CDIM, nullptr, nullptr, nullptr, qkv, 0);
  // 4) causal flash attention
  {
    size_t smem = (size_t)(3 * 128 * 64) * 2 + (size_t)(128 * 128) * 4 + 3 * 128 * 4;
    attn_kernel<<<dim3(TSEQ / 128, 2 * NHEAD), 256, smem, stream>>>(qkv, obf);
  }
  // 5) x1 = x + obf @ Wproj + bproj (f32)
  gemm_bf16<<<dim3(CDIM / 128, MROWS / 128), 256, gemm_smem, stream>>>(
      obf, wproj, CDIM, CDIM, bproj, x, x1, nullptr, 0);
  // 6) LN2
  ln_to_bf16<<<MROWS, 256, 0, stream>>>(x1, g2, be2, h2);
  // 7) ff1 = relu(h2 @ W1 + b1) -> bf16
  gemm_bf16<<<dim3(4096 / 128, MROWS / 128), 256, gemm_smem, stream>>>(
      h2, w1b, 4096, CDIM, b1, nullptr, nullptr, ff1, 1);
  // 8) out = x1 + ff1 @ W2 + b2 -> f32 d_out
  gemm_bf16<<<dim3(CDIM / 128, MROWS / 128), 256, gemm_smem, stream>>>(
      ff1, w2b, CDIM, 4096, b2, x1, (float*)d_out, nullptr, 0);
}
